// EIRNN_61830349193955
// MI455X (gfx1250) — matrix operand order repaired and checked by reading, not compiled, not run on hardware
//
#include <hip/hip_runtime.h>
#include <hip/hip_bf16.h>

// ---------------------------------------------------------------------------
// E-I RNN (Dale's law) forward on MI455X / gfx1250.
//
//   * 4 persistent workgroups, one per 16-row batch tile (recurrence is
//     independent per batch sample), each loops over all T=1000 steps.
//   * Recurrent GEMM via v_wmma_f32_16x16x32_bf16, fp32 accumulate, fp32
//     state x held in VGPRs. 8 waves x 5 N-tiles = 640 output columns.
//   * W_rec^T (Dale-transformed) pre-swizzled into per-lane WMMA B-fragment
//     order -> coalesced 2x global_load_b128 per fragment, L2-resident.
//     The fragment base pointer is laundered through an empty asm each step
//     (blocks LICM hoist + spill seen in round 1) and re-cast into
//     address_space(1) so the loads select global_load_b128, not
//     flat_load_b128 (round 2 emitted FLAT, which needlessly ties LOADcnt
//     to DScnt on CDNA5).
//   * B fragments ping-pong double-buffered across K-tiles; kt=0 prefetch
//     issued before the step barrier to maximize overlap.
//   * Input projection folded in as a 21st K-tile (u_t staged into the
//     padded A matrix in LDS), so the whole update is pure WMMA.
//   * r (bf16) staged in LDS with stride 680 (bank-conflict-free b128 reads).
//   * Readout y via LDS partial sums; rates streamed fp32 to HBM.
//   * Noise: deterministic hash-based ~N(0,1), scale 0.01*sqrt(alpha)
//     (JAX threefry is not bit-reproducible on device).
// ---------------------------------------------------------------------------

typedef __attribute__((ext_vector_type(16))) __bf16 v16bf;
typedef __attribute__((ext_vector_type(8)))  __bf16 v8bf;
typedef __attribute__((ext_vector_type(8)))  float  v8f;

// global (address_space(1)) pointer types for the B-fragment stream
typedef const __attribute__((address_space(1))) __bf16* gbf16p;
typedef const __attribute__((address_space(1))) v16bf*  gv16bfp;

#define N_EXC    512
#define N_TOT    640
#define N_IN     16
#define BATCH    64
#define T_STEPS  1000
#define KT_REC   20          // 640 / 32 recurrent K-tiles
#define KT_ALL   21          // + 1 tile carrying W_in (K 640..655) and zeros
#define NT_ALL   40          // 640 / 16 output-column tiles
#define A_STRIDE 680         // padded: 1360B rows -> 16B aligned, no bank conflicts
#define WAVES    8
#define TPW      5           // N-tiles per wave (8*5 = 40)
#define ALPHA_C  0.5f
#define NOISE_C  0.0070710678f   // 0.01 * sqrt(0.5)

#define FRAG_ELEMS   (32 * 16)            // bf16 elems per B fragment
#define KT_ELEMS     (NT_ALL * FRAG_ELEMS)

// deterministic ~N(0,1): integer hash -> CLT over 4 bytes (var 1/3, scaled)
__device__ __forceinline__ float gnoise(unsigned int idx) {
    unsigned int h = idx * 2654435761u;
    h ^= h >> 16; h *= 0x7feb352du;
    h ^= h >> 15; h *= 0x846ca68bu;
    h ^= h >> 16;
    float s = (float)((h & 255u) + ((h >> 8) & 255u) +
                      ((h >> 16) & 255u) + ((h >> 24) & 255u));
    return (s * (1.0f / 255.0f) - 2.0f) * 1.7320508f;
}

// ---------------------------------------------------------------------------
// Prep: build pre-swizzled bf16 B fragments  wb[kt][nt][lane][e]
//   B[j=K][i=N] = W_rec[i][j] = |raw[i][j]| * sign(j) * (i != j)
//   kt == 20 carries W_in^T (k_local < 16) and zeros.
//   Fragment layout (B 32x16 bf16, wave32): lane<16 -> N=lane, K = e (0..15);
//   lane>=16 -> N=lane-16, K = 16 + e.
// ---------------------------------------------------------------------------
__global__ __launch_bounds__(256) void eirnn_prep_kernel(
    const float* __restrict__ Wrec,   // [640][640]
    const float* __restrict__ Win,    // [640][16]
    __bf16* __restrict__ wb)          // [21][40][32][16]
{
    int idx = blockIdx.x * 256 + threadIdx.x;
    if (idx >= KT_ALL * NT_ALL * 32 * 16) return;
    int e    = idx & 15;
    int lane = (idx >> 4) & 31;
    int tile = idx >> 9;
    int nt   = tile % NT_ALL;
    int kt   = tile / NT_ALL;
    int n    = nt * 16 + (lane & 15);
    int kl   = (lane >> 4) * 16 + e;       // k within the 32-wide tile
    float val;
    if (kt < KT_REC) {
        int k = kt * 32 + kl;
        float raw = Wrec[n * N_TOT + k];
        float sgn = (k >= N_EXC) ? -1.0f : 1.0f;
        val = (n == k) ? 0.0f : fabsf(raw) * sgn;
    } else {
        val = (kl < N_IN) ? Win[n * N_IN + kl] : 0.0f;
    }
    wb[idx] = (__bf16)val;
}

// ---------------------------------------------------------------------------
// Main persistent recurrence kernel: 4 blocks x 256 threads (8 waves).
// ---------------------------------------------------------------------------
__global__ __launch_bounds__(256) void eirnn_step_kernel(
    const float*  __restrict__ inputs,   // [B][T][16]
    const __bf16* __restrict__ wb,       // [21][40][32][16]
    const float*  __restrict__ W_out,    // [2][512]
    const float*  __restrict__ b_out,    // [2]
    float*        __restrict__ out_rates,// [B][T][640]
    float*        __restrict__ out_y)    // [B][T][2]
{
    __shared__ __bf16 sA[16 * A_STRIDE];      // bf16 A matrix (r | u | 0)
    __shared__ float  sWout[2 * N_EXC];
    __shared__ float  sYpart[32 * WAVES];
    __shared__ float  sBout[2];

    const int tid    = threadIdx.x;
    const int lane   = tid & 31;
    const int wave   = tid >> 5;
    const int laneN  = lane & 15;
    const int laneHi = lane >> 4;
    const int bbase  = blockIdx.x * 16;       // batch tile base

    for (int i = tid; i < 2 * N_EXC; i += 256) sWout[i] = W_out[i];
    if (tid < 2) sBout[tid] = b_out[tid];
    // zero trailing A columns [656, 680): K=656..671 is the WMMA zero-pad of
    // the input tile, the rest is bank-padding.
    for (int i = tid; i < 16 * (A_STRIDE - 656); i += 256) {
        int row = i / (A_STRIDE - 656);
        int col = 656 + i % (A_STRIDE - 656);
        sA[row * A_STRIDE + col] = (__bf16)0.0f;
    }

    // per-lane base into the pre-swizzled B fragments
    const __bf16* wbp = wb + ((size_t)(wave * TPW * 32 + lane)) * 16;

    float x[TPW][8];
    #pragma unroll
    for (int i = 0; i < TPW; ++i)
        #pragma unroll
        for (int v = 0; v < 8; ++v) x[i][v] = 0.0f;

    for (int t = 0; t < T_STEPS; ++t) {
        // Launder the B pointer (opaque per t-iteration: blocks LICM/spill),
        // then rebuild it as a GLOBAL address-space pointer so the selected
        // opcode is global_load_b128 (not flat_load_b128).
        unsigned long long bq = (unsigned long long)wbp;
        asm volatile("" : "+v"(bq));
        gbf16p bptr = (gbf16p)bq;

        // Prefetch kt=0 B fragments before the barrier: they do not depend
        // on this step's LDS contents, so their latency hides behind the
        // softplus/staging work and the barrier itself.
        v16bf bfrag[2][TPW];
        #pragma unroll
        for (int i = 0; i < TPW; ++i)
            bfrag[0][i] = *(gv16bfp)(bptr + i * FRAG_ELEMS);

        // ---- 1) r = softplus(x): stage bf16 A matrix + stream rates ------
        #pragma unroll
        for (int i = 0; i < TPW; ++i) {
            int n = wave * (TPW * 16) + i * 16 + laneN;
            #pragma unroll
            for (int v = 0; v < 8; ++v) {
                int m = v + laneHi * 8;
                float xv = x[i][v];
                float rv = fmaxf(xv, 0.0f) + log1pf(__expf(-fabsf(xv)));
                sA[m * A_STRIDE + n] = (__bf16)rv;
                out_rates[((size_t)(bbase + m) * T_STEPS + t) * N_TOT + n] = rv;
            }
        }
        // wave 0 stages u_t (bf16) into A columns 640..655
        if (wave == 0) {
            int m  = laneN;
            int kh = laneHi;                      // which 8-float half
            const float* up =
                inputs + ((size_t)(bbase + m) * T_STEPS + t) * N_IN + kh * 8;
            v8bf tb;
            #pragma unroll
            for (int k = 0; k < 8; ++k) tb[k] = (__bf16)up[k];
            *(v8bf*)&sA[m * A_STRIDE + 640 + kh * 8] = tb;
        }
        __syncthreads();

        // ---- 2) readout partials: lane -> (m, o), wave -> 64-wide n chunk -
        {
            int m = lane >> 1, o = lane & 1;
            const __bf16* rp = &sA[m * A_STRIDE + wave * 64];
            const float*  wp = &sWout[o * N_EXC + wave * 64];
            float acc = 0.0f;
            #pragma unroll
            for (int c = 0; c < 8; ++c) {
                v8bf rv = *(const v8bf*)(rp + c * 8);
                #pragma unroll
                for (int j = 0; j < 8; ++j)
                    acc += (float)rv[j] * wp[c * 8 + j];
            }
            sYpart[lane * WAVES + wave] = acc;
        }

        // ---- 3) rec + ext via WMMA: 21 K-tiles x 5 N-tiles per wave -------
        //        B fragments ping-pong double-buffered across K-tiles.
        v8f acc[TPW];
        #pragma unroll
        for (int i = 0; i < TPW; ++i) acc[i] = (v8f)0.0f;

        #pragma unroll
        for (int kt = 0; kt < KT_ALL; ++kt) {
            const int cur = kt & 1;
            const int nxt = cur ^ 1;
            if (kt + 1 < KT_ALL) {
                gbf16p bn = bptr + (size_t)(kt + 1) * KT_ELEMS;
                #pragma unroll
                for (int i = 0; i < TPW; ++i)
                    bfrag[nxt][i] = *(gv16bfp)(bn + i * FRAG_ELEMS);
            }
            // A fragment from LDS:
            //   lane<16 row=laneN K={0..15}; lane>=16 K={16..31}
            const __bf16* a0 = &sA[laneN * A_STRIDE + kt * 32 + laneHi * 16];
            v8bf lo = *(const v8bf*)a0;          // K base + 0..7
            v8bf hi = *(const v8bf*)(a0 + 8);    // K base + 8..15
            v16bf a = __builtin_shufflevector(lo, hi, 0, 1, 2, 3, 4, 5, 6, 7,
                                              8, 9, 10, 11, 12, 13, 14, 15);
            #pragma unroll
            for (int i = 0; i < TPW; ++i) {
                acc[i] = __builtin_amdgcn_wmma_f32_16x16x32_bf16(
                    false, a, false, bfrag[cur][i], (short)0, acc[i],
                    false, false);
            }
        }

        // ---- 4) leaky state update + injected noise -----------------------
        #pragma unroll
        for (int i = 0; i < TPW; ++i) {
            int n = wave * (TPW * 16) + i * 16 + laneN;
            #pragma unroll
            for (int v = 0; v < 8; ++v) {
                int m = v + laneHi * 8;
                unsigned int idx =
                    ((unsigned int)t * BATCH + (unsigned int)(bbase + m)) * N_TOT + n;
                x[i][v] = (1.0f - ALPHA_C) * x[i][v] + ALPHA_C * acc[i][v] +
                          NOISE_C * gnoise(idx);
            }
        }
        __syncthreads();   // sA/sYpart consumed; safe to overwrite next step

        // ---- 5) wave 0 finalizes y ---------------------------------------
        if (wave == 0) {
            float s = sBout[lane & 1];
            #pragma unroll
            for (int w = 0; w < WAVES; ++w) s += sYpart[lane * WAVES + w];
            int b = bbase + (lane >> 1);
            out_y[((size_t)b * T_STEPS + t) * 2 + (lane & 1)] = s;
        }
    }
}

// ---------------------------------------------------------------------------
extern "C" void kernel_launch(void* const* d_in, const int* in_sizes, int n_in,
                              void* d_out, int out_size, void* d_ws, size_t ws_size,
                              hipStream_t stream) {
    const float* inputs    = (const float*)d_in[0];  // [64][1000][16]
    const float* W_rec_raw = (const float*)d_in[1];  // [640][640]
    const float* W_in      = (const float*)d_in[2];  // [640][16]
    const float* W_out     = (const float*)d_in[3];  // [2][512]
    const float* b_out     = (const float*)d_in[4];  // [2]

    float* out       = (float*)d_out;
    float* out_rates = out;                                       // [64][1000][640]
    float* out_y     = out + (size_t)BATCH * T_STEPS * N_TOT;     // [64][1000][2]

    __bf16* wb = (__bf16*)d_ws;   // 21*40*32*16 bf16 = 860,160 B of scratch

    int prep_elems  = KT_ALL * NT_ALL * 32 * 16;
    int prep_blocks = (prep_elems + 255) / 256;
    eirnn_prep_kernel<<<prep_blocks, 256, 0, stream>>>(W_rec_raw, W_in, wb);
    eirnn_step_kernel<<<BATCH / 16, 256, 0, stream>>>(
        inputs, wb, W_out, b_out, out_rates, out_y);
}
